// LSTMKerasModel_12378095747939
// MI455X (gfx1250) — compile-verified, matrix operand order checked by reading, not run on hardware
//
#include <hip/hip_runtime.h>
#include <hip/hip_bf16.h>
#include <math.h>

typedef __bf16 bf16;
typedef __attribute__((ext_vector_type(16))) __bf16 v16bf;
typedef __attribute__((ext_vector_type(8)))  float  v8f;

// Problem constants
#define BB   64
#define TT   1024
#define FF   256
#define HH   512
#define G4H  2048   // 4*H

// ---------------------------------------------------------------------------
// Fragment element -> K mapping for 16-bit A/B WMMA fragments (ISA 7.12.2):
//   lanes 0-15 hold K offsets {0..7, 16..23}, lanes 16-31 hold {8..15, 24..31}
//   element e (0..15):  K = e + (e>=8 ? 8 : 0) + (lane>=16 ? 8 : 0)
// ---------------------------------------------------------------------------

__device__ __forceinline__ float fast_sigmoid(float x) {
    return 1.0f / (1.0f + __expf(-x));
}
__device__ __forceinline__ float fast_tanh(float x) {
    return 2.0f / (1.0f + __expf(-2.0f * x)) - 1.0f;
}

// Pack a [Krows, 2048] f32 weight matrix (row-major, K outer) into bf16
// B-fragment layout: dst[((nt*Ktiles + kt)*32 + lane)*16 + e]
__global__ __launch_bounds__(256) void pack_B(const float* __restrict__ src,
                                              bf16* __restrict__ dst, int Ktiles) {
    int fid  = blockIdx.x * blockDim.x + threadIdx.x;   // (nt*Ktiles + kt)*32 + lane
    int lane = fid & 31;
    int kt   = (fid >> 5) % Ktiles;
    int nt   = fid / (32 * Ktiles);
    int hi8  = (lane >= 16) ? 8 : 0;
    int n    = nt * 16 + (lane & 15);
    bf16* o  = dst + (size_t)fid * 16;
#pragma unroll
    for (int e = 0; e < 16; ++e) {
        int k = kt * 32 + e + ((e >= 8) ? 8 : 0) + hi8;
        o[e]  = (bf16)src[(size_t)k * G4H + n];
    }
}

// Pack inputs [B, T, F] f32 into per-timestep A-fragment layout:
// dst[(((t*4 + mt)*8 + kt)*32 + lane)*16 + e]
__global__ __launch_bounds__(256) void pack_X(const float* __restrict__ src,
                                              bf16* __restrict__ dst) {
    int fid  = blockIdx.x * blockDim.x + threadIdx.x;   // t*1024 + mt*256 + kt*32 + lane
    int lane = fid & 31;
    int kt   = (fid >> 5) & 7;
    int mt   = (fid >> 8) & 3;
    int t    = fid >> 10;
    int hi8  = (lane >= 16) ? 8 : 0;
    int brow = mt * 16 + (lane & 15);
    const float* s = src + (size_t)brow * TT * FF + (size_t)t * FF;
    bf16* o = dst + (size_t)fid * 16;
#pragma unroll
    for (int e = 0; e < 16; ++e) {
        int f = kt * 32 + e + ((e >= 8) ? 8 : 0) + hi8;
        o[e]  = (bf16)s[f];
    }
}

// Precompute Zx[t] = x_t @ W + b for ALL timesteps (fully parallel, fills the
// device). Output layout: per-wave-tile f32 D fragments:
//   Zx[((((t*4+mt)*32 + jt)*4 + gi)*32 + lane)*8 + r]
__global__ __launch_bounds__(128) void xw_gemm(const bf16*  __restrict__ Xp,
                                               const bf16*  __restrict__ Wp,
                                               const float* __restrict__ bias,
                                               float*       __restrict__ Zx) {
    const int lane = threadIdx.x & 31;
    const int wid  = (blockIdx.x * blockDim.x + threadIdx.x) >> 5;  // 0 .. T*128-1
    const int t    = wid >> 7;
    const int mt   = (wid >> 5) & 3;
    const int jt   = wid & 31;

    v8f acc[4] = {v8f{}, v8f{}, v8f{}, v8f{}};
    const bf16* xa = Xp + ((size_t)((t * 4 + mt) * 8) * 32 + lane) * 16;
    for (int kt = 0; kt < 8; ++kt) {
        v16bf a = *(const v16bf*)(xa + (size_t)kt * 32 * 16);
#pragma unroll
        for (int gi = 0; gi < 4; ++gi) {
            const bf16* bp = Wp + (((size_t)(gi * 32 + jt) * 8 + kt) * 32 + lane) * 16;
            v16bf bfrag = *(const v16bf*)bp;
            acc[gi] = __builtin_amdgcn_wmma_f32_16x16x32_bf16(
                false, a, false, bfrag, (short)0, acc[gi], false, false);
        }
    }
    const int n = jt * 16 + (lane & 15);
#pragma unroll
    for (int gi = 0; gi < 4; ++gi) {
        float bv = bias[gi * HH + n];
#pragma unroll
        for (int r = 0; r < 8; ++r) acc[gi][r] += bv;
        float* dst = Zx + (((((size_t)t * 4 + mt) * 32 + jt) * 4 + gi) * 32 + lane) * 8;
        *(v8f*)dst = acc[gi];
    }
}

// One LSTM timestep. Grid: 32 blocks x 128 threads = 128 waves.
// Block b: mt = b>>3, jt = (b&7)*4 + warp. Each wave owns one 16x16 tile of
// all four gates, then the block repacks its h tiles into bf16 A-fragment
// layout (hA) via LDS for the next step.
// launch_bounds(128, 1): one block per CU -> allow a deep register pipeline.
template <bool USE_ZX>
__global__ __launch_bounds__(128, 1) void lstm_step(
    const float* __restrict__ Zx,     // f32 x-projection fragments (USE_ZX)
    const bf16*  __restrict__ Xp,     // packed input A-fragments (!USE_ZX)
    const bf16*  __restrict__ Wp,     // packed W B-fragments (!USE_ZX)
    const bf16*  __restrict__ Up,     // packed U B-fragments (Ktiles=16)
    const float* __restrict__ bias,   // [4H] (added here only if !USE_ZX)
    const bf16*  __restrict__ hA_in,  // packed h A-fragments (prev step)
    bf16*        __restrict__ hA_out, // packed h A-fragments (this step)
    float*       __restrict__ h_f32,  // [B, H] hidden (f32, for head)
    float*       __restrict__ c,      // [B, H] cell state (in-place)
    int t)
{
    __shared__ float tile[16][65];    // block's 16 x 64 h-tile (padded)

    const int lane = threadIdx.x & 31;
    const int warp = threadIdx.x >> 5;            // 0..3
    const int mt   = blockIdx.x >> 3;             // 0..3
    const int jt   = (blockIdx.x & 7) * 4 + warp; // 0..31

    // Issue ALL recurrent A-fragment loads up front (critical cross-step dep):
    // 16 fragments x 8 VGPRs; one long clause stream instead of 16 stalls.
    v16bf ah[16];
    {
        const bf16* ha = hA_in + ((size_t)(mt * 16) * 32 + lane) * 16;
#pragma unroll
        for (int kt = 0; kt < 16; ++kt)
            ah[kt] = *(const v16bf*)(ha + (size_t)kt * 32 * 16);
    }

    v8f acc[4];
    if (USE_ZX) {
        const float* zp = Zx + ((((size_t)t * 4 + mt) * 32 + jt) * 4) * 32 * 8 + (size_t)lane * 8;
#pragma unroll
        for (int gi = 0; gi < 4; ++gi)
            acc[gi] = *(const v8f*)(zp + (size_t)gi * 32 * 8);
    } else {
#pragma unroll
        for (int gi = 0; gi < 4; ++gi) acc[gi] = v8f{};
        // Phase 1: x_t @ W  (K = 256, 8 k-tiles)
        const bf16* xa = Xp + ((size_t)((t * 4 + mt) * 8) * 32 + lane) * 16;
#pragma unroll
        for (int kt = 0; kt < 8; ++kt) {
            v16bf a = *(const v16bf*)(xa + (size_t)kt * 32 * 16);
#pragma unroll
            for (int gi = 0; gi < 4; ++gi) {
                const bf16* bp = Wp + (((size_t)(gi * 32 + jt) * 8 + kt) * 32 + lane) * 16;
                v16bf bfrag = *(const v16bf*)bp;
                acc[gi] = __builtin_amdgcn_wmma_f32_16x16x32_bf16(
                    false, a, false, bfrag, (short)0, acc[gi], false, false);
            }
        }
    }

    // Phase 2: h_prev @ U  (K = 512, 16 k-tiles). Fully unrolled so B-fragment
    // loads (compile-time offsets from one base) can be hoisted ahead of use.
    {
        const bf16* ub = Up + ((size_t)jt * 16 * 32 + lane) * 16;
#pragma unroll
        for (int kt = 0; kt < 16; ++kt) {
#pragma unroll
            for (int gi = 0; gi < 4; ++gi) {
                const bf16* bp = ub + ((size_t)gi * 32 * 16 + (size_t)kt) * 32 * 16;
                v16bf bfrag = *(const v16bf*)bp;
                acc[gi] = __builtin_amdgcn_wmma_f32_16x16x32_bf16(
                    false, ah[kt], false, bfrag, (short)0, acc[gi], false, false);
            }
        }
    }

    // Gates + state update (D layout: lane owns column n, rows r / r+8)
    const int n = jt * 16 + (lane & 15);
    float bi = 0.f, bf_ = 0.f, bg = 0.f, bo = 0.f;
    if (!USE_ZX) {
        bi  = bias[0 * HH + n];
        bf_ = bias[1 * HH + n];
        bg  = bias[2 * HH + n];
        bo  = bias[3 * HH + n];
    }
    const int rhi   = (lane >= 16) ? 8 : 0;
    const int mbase = mt * 16 + rhi;
    const int lcol  = warp * 16 + (lane & 15);    // column within block tile
#pragma unroll
    for (int r = 0; r < 8; ++r) {
        size_t idx = (size_t)(mbase + r) * HH + n;
        float iv = fast_sigmoid(acc[0][r] + bi);
        float fv = fast_sigmoid(acc[1][r] + bf_);
        float gv = fast_tanh   (acc[2][r] + bg);
        float ov = fast_sigmoid(acc[3][r] + bo);
        float cn = fv * c[idx] + iv * gv;
        float hn = ov * fast_tanh(cn);
        c[idx]     = cn;
        h_f32[idx] = hn;
        tile[rhi + r][lcol] = hn;
    }
    __syncthreads();

    // Repack block's 16x64 h tile into two bf16 A-fragments (k-tiles over H).
    // Block covers H columns [ (blockIdx&7)*64 , +64 ) = hA k-tiles 2b, 2b+1.
    if (threadIdx.x < 64) {
        const int w2  = threadIdx.x >> 5;             // 0..1 : which k-tile
        const int l   = threadIdx.x & 31;
        const int kt  = (blockIdx.x & 7) * 2 + w2;    // hA k-tile index
        const int row = l & 15;
        const int h8  = (l >= 16) ? 8 : 0;
        bf16* o = hA_out + (((size_t)(mt * 16 + kt)) * 32 + l) * 16;
#pragma unroll
        for (int e = 0; e < 16; ++e) {
            int col = w2 * 32 + e + ((e >= 8) ? 8 : 0) + h8;  // K(e, l) within 64
            o[e] = (bf16)tile[row][col];
        }
    }
}

// out[b] = dot(h[b,:], Wd) + bd. Grid: 64 blocks x 256 threads.
__global__ __launch_bounds__(256) void head_kernel(const float* __restrict__ h,
                                                   const float* __restrict__ Wd,
                                                   const float* __restrict__ bd,
                                                   float* __restrict__ out) {
    __shared__ float s[256];
    const int b = blockIdx.x, tid = threadIdx.x;
    float v = h[(size_t)b * HH + tid] * Wd[tid]
            + h[(size_t)b * HH + tid + 256] * Wd[tid + 256];
    s[tid] = v;
    __syncthreads();
    for (int off = 128; off > 0; off >>= 1) {
        if (tid < off) s[tid] += s[tid + off];
        __syncthreads();
    }
    if (tid == 0) out[b] = s[0] + bd[0];
}

extern "C" void kernel_launch(void* const* d_in, const int* in_sizes, int n_in,
                              void* d_out, int out_size, void* d_ws, size_t ws_size,
                              hipStream_t stream) {
    (void)in_sizes; (void)n_in; (void)out_size;
    const float* inputs = (const float*)d_in[0];  // [B, T, F]
    const float* W      = (const float*)d_in[1];  // [F, 4H]
    const float* U      = (const float*)d_in[2];  // [H, 4H]
    const float* bias   = (const float*)d_in[3];  // [4H]
    const float* Wd     = (const float*)d_in[4];  // [H, 1]
    const float* bd     = (const float*)d_in[5];  // [1]
    float* out = (float*)d_out;                   // [B, 1]

    // Workspace carve-up (all chunk sizes multiples of 4 KiB)
    const size_t WP_B  = (size_t)128 *  8 * 32 * 16 * sizeof(bf16);   // 1 MiB
    const size_t UP_B  = (size_t)128 * 16 * 32 * 16 * sizeof(bf16);   // 2 MiB
    const size_t XP_B  = (size_t)TT * 4 * 8 * 32 * 16 * sizeof(bf16); // 32 MiB
    const size_t C_B   = (size_t)BB * HH * sizeof(float);             // 128 KiB
    const size_t HA_B  = (size_t)4 * 16 * 32 * 16 * sizeof(bf16);     // 64 KiB
    const size_t H_B   = (size_t)BB * HH * sizeof(float);             // 128 KiB
    const size_t ZX_B  = (size_t)TT * 4 * 32 * 4 * 32 * 8 * sizeof(float); // 512 MiB

    char* p = (char*)d_ws;
    bf16*  Wp  = (bf16*)p;  p += WP_B;
    bf16*  Up  = (bf16*)p;  p += UP_B;
    bf16*  Xp  = (bf16*)p;  p += XP_B;
    float* c   = (float*)p; p += C_B;
    bf16*  hA0 = (bf16*)p;  p += HA_B;
    bf16*  hA1 = (bf16*)p;  p += HA_B;
    float* h   = (float*)p; p += H_B;
    const size_t base_bytes = (size_t)(p - (char*)d_ws);
    const bool use_zx = (ws_size >= base_bytes + ZX_B);
    float* Zx = (float*)p;   // valid only if use_zx

    // Zero c, hA0, hA1 (contiguous block)
    hipMemsetAsync(c, 0, C_B + 2 * HA_B, stream);

    // Pack weights and inputs into WMMA fragment layouts (bf16)
    pack_B<<<128, 256, 0, stream>>>(W, Wp, 8);    // 128 Ntiles * 8 Ktiles * 32 lanes
    pack_B<<<256, 256, 0, stream>>>(U, Up, 16);   // 128 Ntiles * 16 Ktiles * 32 lanes
    pack_X<<<4096, 256, 0, stream>>>(inputs, Xp); // T * 4 * 8 * 32 lanes

    if (use_zx) {
        // Big parallel GEMM: Zx = X @ W + b for all timesteps (T*128 waves)
        xw_gemm<<<TT * 32, 128, 0, stream>>>(Xp, Wp, bias, Zx);
        for (int t = 0; t < TT; ++t) {
            const bf16* hin  = (t & 1) ? hA1 : hA0;
            bf16*       hout = (t & 1) ? hA0 : hA1;
            lstm_step<true><<<32, 128, 0, stream>>>(Zx, Xp, Wp, Up, bias,
                                                    hin, hout, h, c, t);
        }
    } else {
        for (int t = 0; t < TT; ++t) {
            const bf16* hin  = (t & 1) ? hA1 : hA0;
            bf16*       hout = (t & 1) ? hA0 : hA1;
            lstm_step<false><<<32, 128, 0, stream>>>(nullptr, Xp, Wp, Up, bias,
                                                     hin, hout, h, c, t);
        }
    }
    head_kernel<<<BB, 256, 0, stream>>>(h, Wd, bd, out);
}
